// PerformancePredictorGNN_12747462934560
// MI455X (gfx1250) — matrix operand order repaired and checked by reading, not compiled, not run on hardware
//
#include <hip/hip_runtime.h>
#include <hip/hip_bf16.h>
#include <cstdint>

typedef __attribute__((ext_vector_type(16))) _Float16 v16h;
typedef __attribute__((ext_vector_type(8)))  _Float16 v8h;
typedef __attribute__((ext_vector_type(8)))  float    v8f;

#define N_NODES 100000
#define N_EDGES 1600000
#define D_EMB   96
#define D_H     128
#define D_MLP   256
#define N_B     64

// ---------------------------------------------------------------------------
// degree / norm
// ---------------------------------------------------------------------------
__global__ void k_deg(const int* __restrict__ col, float* __restrict__ deg, int nE) {
    int e = blockIdx.x * blockDim.x + threadIdx.x;
    if (e >= nE) return;
    atomicAdd(&deg[col[e]], 1.0f);
}

__global__ void k_dinv(float* __restrict__ d, int n) {
    int i = blockIdx.x * blockDim.x + threadIdx.x;
    if (i < n) d[i] = rsqrtf(d[i] + 1.0f);   // +1 = self loop
}

// ---------------------------------------------------------------------------
// embedding gather -> f16
// ---------------------------------------------------------------------------
__global__ void k_embed(const int* __restrict__ x, const float* __restrict__ emb,
                        _Float16* __restrict__ out, int total) {
    int i = blockIdx.x * blockDim.x + threadIdx.x;
    if (i >= total) return;
    int n = i / D_EMB, d = i - n * D_EMB;
    out[i] = (_Float16)emb[x[n] * D_EMB + d];
}

// W [K][Ncol] f32 row-major  ->  Wt [Ncol][K] f16 row-major (transposed)
__global__ void k_transpose(const float* __restrict__ W, _Float16* __restrict__ Wt,
                            int K, int Ncol) {
    int i = blockIdx.x * blockDim.x + threadIdx.x;
    if (i >= K * Ncol) return;
    int n = i / K, k = i - n * K;
    Wt[i] = (_Float16)W[k * Ncol + n];
}

// ---------------------------------------------------------------------------
// WMMA GEMM:  C[N x 128] = A[N x K] (f16) * Bt^T  (Bt is [128][K] f16)
// block = 256 threads = 8 waves; block owns 16 rows, wave owns 16 cols.
// ---------------------------------------------------------------------------
__global__ void __launch_bounds__(256)
k_gemm(const _Float16* __restrict__ A, const _Float16* __restrict__ Bt,
       const float* __restrict__ bias, float* __restrict__ C, int K, int nRows) {
    __shared__ _Float16 tA[16 * 128];          // 16 rows x K<=128 halves

    int r0 = blockIdx.x * 16;
    if (r0 >= nRows) return;

    // cooperative contiguous copy of the 16xK tile (rows are contiguous, lda=K)
    const uint32_t* srcA = (const uint32_t*)(A + (size_t)r0 * K);
    uint32_t* dA = (uint32_t*)tA;
    int totDw = 8 * K;                          // 16*K halves / 2
    for (int i = threadIdx.x; i < totDw; i += 256) dA[i] = srcA[i];
    __syncthreads();

    int waveId = threadIdx.x >> 5;              // 0..7 -> column tile
    int lane   = threadIdx.x & 31;
    int sub    = lane & 15;
    int sel    = lane >> 4;                     // 0/1 half-wave select
    int c0     = waveId * 16;

    v8f acc = {};
    for (int k0 = 0; k0 < K; k0 += 32) {
        // A fragment (16-bit A 16x32 layout): chunks at k0+sel*8 and +16
        const _Float16* pa = tA + sub * K + k0 + sel * 8;
        v8h alo = *(const v8h*)pa;
        v8h ahi = *(const v8h*)(pa + 16);
        v16h a;
#pragma unroll
        for (int i = 0; i < 8; ++i) { a[i] = alo[i]; a[i + 8] = ahi[i]; }

        // B fragment: lane holds column (c0+sub); chunks at k0+sel*8 and +16
        const _Float16* pb = Bt + (size_t)(c0 + sub) * K + k0 + sel * 8;
        v8h blo = *(const v8h*)pb;
        v8h bhi = *(const v8h*)(pb + 16);
        v16h b;
#pragma unroll
        for (int i = 0; i < 8; ++i) { b[i] = blo[i]; b[i + 8] = bhi[i]; }

        acc = __builtin_amdgcn_wmma_f32_16x16x32_f16(false, a, false, b,
                                                     (short)0, acc, false, false);
    }

    float bv = bias ? bias[c0 + sub] : 0.0f;
#pragma unroll
    for (int v = 0; v < 8; ++v) {
        int rr = r0 + v + sel * 8;              // C/D layout: M = v + 8*sel
        C[(size_t)rr * D_H + c0 + sub] = acc[v] + bv;
    }
}

// ---------------------------------------------------------------------------
// edge scatter: agg[col] += m[row] * dinv[row]*dinv[col]   (wave per edge)
// ---------------------------------------------------------------------------
__global__ void __launch_bounds__(256)
k_scatter(const int* __restrict__ row, const int* __restrict__ col,
          const float* __restrict__ dinv, const float* __restrict__ m,
          float* __restrict__ agg, int nE) {
    int e    = (blockIdx.x * blockDim.x + threadIdx.x) >> 5;
    int lane = threadIdx.x & 31;
    if (e >= nE) return;
    int r = row[e], c = col[e];
    float w = dinv[r] * dinv[c];
    float4 v = ((const float4*)(m + (size_t)r * D_H))[lane];
    float* dst = agg + (size_t)c * D_H + lane * 4;
    atomicAdd(dst + 0, v.x * w);
    atomicAdd(dst + 1, v.y * w);
    atomicAdd(dst + 2, v.z * w);
    atomicAdd(dst + 3, v.w * w);
}

// ---------------------------------------------------------------------------
// combine: h = 0.5*skip + 0.5*(agg + m*dinv^2 + bias) [+relu]; optional f16 copy
// ---------------------------------------------------------------------------
__global__ void k_combine(const float* __restrict__ skip, const float* __restrict__ agg,
                          const float* __restrict__ m, const float* __restrict__ dinv,
                          const float* __restrict__ bias, float* __restrict__ hOut,
                          _Float16* __restrict__ h16Out, int doRelu, int total) {
    int i = blockIdx.x * blockDim.x + threadIdx.x;
    if (i >= total) return;
    int n = i >> 7, d = i & (D_H - 1);
    float di = dinv[n];
    float hf = agg[i] + m[i] * di * di + bias[d];
    float h  = 0.5f * skip[i] + 0.5f * hf;
    if (doRelu) h = fmaxf(h, 0.0f);
    hOut[i] = h;
    if (h16Out) h16Out[i] = (_Float16)h;
}

// ---------------------------------------------------------------------------
// connected-node mask (original edges only)
// ---------------------------------------------------------------------------
__global__ void k_conn(const int* __restrict__ row, const int* __restrict__ col,
                       int* __restrict__ conn, int nE) {
    int e = blockIdx.x * blockDim.x + threadIdx.x;
    if (e >= nE) return;
    conn[row[e]] = 1;
    conn[col[e]] = 1;
}

// pooling: wave per node
__global__ void __launch_bounds__(256)
k_pool(const float* __restrict__ h, const int* __restrict__ conn,
       const int* __restrict__ batch, float* __restrict__ sums,
       float* __restrict__ cnt, int nNodes) {
    int node = (blockIdx.x * blockDim.x + threadIdx.x) >> 5;
    int lane = threadIdx.x & 31;
    if (node >= nNodes) return;
    if (!conn[node]) return;
    int b = batch[node];
    float4 v = ((const float4*)(h + (size_t)node * D_H))[lane];
    float* dst = sums + (size_t)b * D_H + lane * 4;
    atomicAdd(dst + 0, v.x);
    atomicAdd(dst + 1, v.y);
    atomicAdd(dst + 2, v.z);
    atomicAdd(dst + 3, v.w);
    if (lane == 0) atomicAdd(&cnt[b], 1.0f);
}

// ---------------------------------------------------------------------------
// readout MLP: out[b] = relu(g@M1+mb1)@M2 + mb2 ; one block per graph
// ---------------------------------------------------------------------------
__global__ void __launch_bounds__(256)
k_mlp(const float* __restrict__ sums, const float* __restrict__ cnt,
      const float* __restrict__ M1, const float* __restrict__ mb1,
      const float* __restrict__ M2, const float* __restrict__ mb2,
      float* __restrict__ out) {
    __shared__ float g[D_H];
    __shared__ float red[256];
    int b = blockIdx.x;
    float c = fmaxf(cnt[b], 1.0f);
    if (threadIdx.x < D_H) g[threadIdx.x] = sums[(size_t)b * D_H + threadIdx.x] / c;
    __syncthreads();
    int j = threadIdx.x;                        // 0..255 hidden units
    float t = mb1[j];
    for (int k = 0; k < D_H; ++k) t += g[k] * M1[k * D_MLP + j];
    t = fmaxf(t, 0.0f);
    red[j] = t * M2[j];
    __syncthreads();
    for (int s = 128; s > 0; s >>= 1) {
        if (j < s) red[j] += red[j + s];
        __syncthreads();
    }
    if (j == 0) out[b] = red[0] + mb2[0];
}

// ---------------------------------------------------------------------------
extern "C" void kernel_launch(void* const* d_in, const int* in_sizes, int n_in,
                              void* d_out, int out_size, void* d_ws, size_t ws_size,
                              hipStream_t stream) {
    const int*   x     = (const int*)d_in[0];
    const int*   ei    = (const int*)d_in[1];
    const int*   row   = ei;
    const int*   col   = ei + N_EDGES;
    const int*   batch = (const int*)d_in[2];
    const float* emb   = (const float*)d_in[3];
    const float* W0 = (const float*)d_in[4];  const float* b0  = (const float*)d_in[5];
    const float* W1 = (const float*)d_in[6];  const float* b1  = (const float*)d_in[7];
    const float* W2 = (const float*)d_in[8];  const float* b2  = (const float*)d_in[9];
    const float* P0 = (const float*)d_in[10]; const float* pb0 = (const float*)d_in[11];
    const float* M1 = (const float*)d_in[12]; const float* mb1 = (const float*)d_in[13];
    const float* M2 = (const float*)d_in[14]; const float* mb2 = (const float*)d_in[15];
    float* out = (float*)d_out;

    // carve workspace
    char* wp = (char*)d_ws;
    auto carve = [&](size_t bytes) {
        void* p = (void*)wp;
        wp += (bytes + 255) & ~(size_t)255;
        return p;
    };
    float*    bufM    = (float*)   carve((size_t)N_NODES * D_H * 4);
    float*    bufP    = (float*)   carve((size_t)N_NODES * D_H * 4);   // skip / h (f32)
    float*    bufAgg  = (float*)   carve((size_t)N_NODES * D_H * 4);
    _Float16* h16emb  = (_Float16*)carve((size_t)N_NODES * D_EMB * 2);
    _Float16* h16     = (_Float16*)carve((size_t)N_NODES * D_H * 2);
    float*    dinv    = (float*)   carve((size_t)N_NODES * 4);
    int*      conn    = (int*)     carve((size_t)N_NODES * 4);
    float*    sums    = (float*)   carve((size_t)N_B * D_H * 4);
    float*    cnt     = (float*)   carve((size_t)N_B * 4);
    _Float16* Wt0     = (_Float16*)carve((size_t)D_EMB * D_H * 2);
    _Float16* Pt0     = (_Float16*)carve((size_t)D_EMB * D_H * 2);
    _Float16* Wt1     = (_Float16*)carve((size_t)D_H * D_H * 2);
    _Float16* Wt2     = (_Float16*)carve((size_t)D_H * D_H * 2);

    const int TB = 256;
    int gE    = (N_EDGES + TB - 1) / TB;        // thread per edge
    int gN    = (N_NODES + TB - 1) / TB;
    int gEmb  = ((size_t)N_NODES * D_EMB + TB - 1) / TB;
    int gElem = ((size_t)N_NODES * D_H + TB - 1) / TB;
    int gGemm = N_NODES / 16;                   // 6250, exact
    int gScat = N_EDGES / 8;                    // wave/edge, 8 edges per block
    int gPool = N_NODES / 8;                    // wave/node

    // 1) degree -> dinv
    hipMemsetAsync(dinv, 0, (size_t)N_NODES * 4, stream);
    k_deg <<<gE, TB, 0, stream>>>(col, dinv, N_EDGES);
    k_dinv<<<gN, TB, 0, stream>>>(dinv, N_NODES);

    // 2) embedding gather (f16) + weight transposes (f16)
    k_embed<<<gEmb, TB, 0, stream>>>(x, emb, h16emb, N_NODES * D_EMB);
    k_transpose<<<(D_EMB * D_H + TB - 1) / TB, TB, 0, stream>>>(W0, Wt0, D_EMB, D_H);
    k_transpose<<<(D_EMB * D_H + TB - 1) / TB, TB, 0, stream>>>(P0, Pt0, D_EMB, D_H);
    k_transpose<<<(D_H  * D_H + TB - 1) / TB, TB, 0, stream>>>(W1, Wt1, D_H, D_H);
    k_transpose<<<(D_H  * D_H + TB - 1) / TB, TB, 0, stream>>>(W2, Wt2, D_H, D_H);

    // ---- layer 0 ----
    k_gemm<<<gGemm, TB, 0, stream>>>(h16emb, Wt0, nullptr, bufM, D_EMB, N_NODES);
    k_gemm<<<gGemm, TB, 0, stream>>>(h16emb, Pt0, pb0,     bufP, D_EMB, N_NODES);
    hipMemsetAsync(bufAgg, 0, (size_t)N_NODES * D_H * 4, stream);
    k_scatter<<<gScat, TB, 0, stream>>>(row, col, dinv, bufM, bufAgg, N_EDGES);
    k_combine<<<gElem, TB, 0, stream>>>(bufP, bufAgg, bufM, dinv, b0,
                                        bufP, h16, 1, N_NODES * D_H);

    // ---- layer 1 ----
    k_gemm<<<gGemm, TB, 0, stream>>>(h16, Wt1, nullptr, bufM, D_H, N_NODES);
    hipMemsetAsync(bufAgg, 0, (size_t)N_NODES * D_H * 4, stream);
    k_scatter<<<gScat, TB, 0, stream>>>(row, col, dinv, bufM, bufAgg, N_EDGES);
    k_combine<<<gElem, TB, 0, stream>>>(bufP, bufAgg, bufM, dinv, b1,
                                        bufP, h16, 1, N_NODES * D_H);

    // ---- layer 2 (no relu, no f16 copy needed) ----
    k_gemm<<<gGemm, TB, 0, stream>>>(h16, Wt2, nullptr, bufM, D_H, N_NODES);
    hipMemsetAsync(bufAgg, 0, (size_t)N_NODES * D_H * 4, stream);
    k_scatter<<<gScat, TB, 0, stream>>>(row, col, dinv, bufM, bufAgg, N_EDGES);
    k_combine<<<gElem, TB, 0, stream>>>(bufP, bufAgg, bufM, dinv, b2,
                                        bufP, (_Float16*)nullptr, 0, N_NODES * D_H);

    // ---- masked mean pool + MLP ----
    hipMemsetAsync(conn, 0, (size_t)N_NODES * 4, stream);
    k_conn<<<gE, TB, 0, stream>>>(row, col, conn, N_EDGES);
    hipMemsetAsync(sums, 0, (size_t)N_B * D_H * 4, stream);
    hipMemsetAsync(cnt,  0, (size_t)N_B * 4, stream);
    k_pool<<<gPool, TB, 0, stream>>>(bufP, conn, batch, sums, cnt, N_NODES);
    k_mlp<<<N_B, TB, 0, stream>>>(sums, cnt, M1, mb1, M2, mb2, out);
}
